// STGNN_MPGNN_NODE_GLOBAL_36060545417512
// MI455X (gfx1250) — compile-verified, matrix operand order checked by reading, not run on hardware
//
#include <hip/hip_runtime.h>
#include <hip/hip_bf16.h>
#include <math.h>

// ---------------------------------------------------------------------------
// STGNN forward for MI455X (gfx1250, wave32, WMMA f32_16x16x32_f16)
// f16 activations everywhere (f32 accumulate), f32 atomic segment-sum.
// ---------------------------------------------------------------------------

typedef _Float16 half_t;
typedef __attribute__((ext_vector_type(16))) _Float16 v16h;
typedef __attribute__((ext_vector_type(8)))  _Float16 v8h;
typedef __attribute__((ext_vector_type(8)))  float    v8f;

#define BB     8
#define TT     128
#define NNODE  16
#define NEDGE  240
#define GNUM   (BB*TT)        // 1024 graphs
#define DD     64
#define NTOT   (GNUM*NNODE)   // 16384 nodes
#define ETOT   (GNUM*NEDGE)   // 245760 edges

__device__ __forceinline__ float leakyf(float v){ return v > 0.f ? v : 0.01f*v; }
__device__ __forceinline__ float sigmf (float v){ return 1.f/(1.f+__expf(-v)); }

// --- WMMA fragment loaders (CDNA5 16x16x32 f16 layout, ISA 7.12.2) ---------
// lane: lm = lane&15 (m for A, n for B), lh = lane>>4
// a[0..7]  <-> k = kbase + 8*lh + 0..7   (contiguous!)
// a[8..15] <-> k = kbase + 16 + 8*lh + 0..7 (contiguous!)

__device__ __forceinline__ v16h load_frag_h(const half_t* rowp, int kbase, int lh){
  const half_t* p = rowp + kbase + lh*8;
  v8h x0 = *(const v8h*)(p);        // 16B load
  v8h x1 = *(const v8h*)(p + 16);   // 16B load
  v16h a;
#pragma unroll
  for (int i = 0; i < 8; ++i) { a[i] = x0[i]; a[8+i] = x1[i]; }
  return a;
}

template<int KA>
__device__ __forceinline__ v16h load_frag_f(const float* rowp, int kbase, int lh){
  v16h a;
#pragma unroll
  for (int i = 0; i < 8; ++i) {
    int k  = kbase + lh*8 + i;
    int k2 = k + 16;
    a[i]   = (_Float16)((k  < KA) ? rowp[k]  : 0.f);
    a[8+i] = (_Float16)((k2 < KA) ? rowp[k2] : 0.f);
  }
  return a;
}

__device__ __forceinline__ v8f wmma32(v16h a, v16h b, v8f c){
  return __builtin_amdgcn_wmma_f32_16x16x32_f16(false, a, false, b, (short)0, c, false, false);
}

// --- weight convert: f32 [K][N] row-major  ->  f16 [N][Kp] (zero padded) ---
__global__ void wcvt_kernel(const float* __restrict__ W, half_t* __restrict__ Wt,
                            int K, int N, int Kp){
  int idx = blockIdx.x * blockDim.x + threadIdx.x;
  if (idx >= N * Kp) return;
  int n = idx / Kp, k = idx - n * Kp;
  float v = (k < K) ? W[(size_t)k * N + n] : 0.f;
  Wt[idx] = (half_t)v;
}

// --- generic GEMM: C[M x NOUT] = act(A[M x KA](f32) @ W + bias) ------------
// KA/KP compile-time -> guards fold, A loads vectorize. OutT = float|half_t.
template<int KA, int KP, typename OutT>
__global__ __launch_bounds__(128) void gemm_wmma(
    const float* __restrict__ A, int lda,
    const half_t* __restrict__ Wt,
    const float* __restrict__ bias,
    OutT* __restrict__ C, int ldc, int M, int act)
{
  int lane = threadIdx.x & 31, wave = threadIdx.x >> 5;
  int lm = lane & 15, lh = lane >> 4;
  int row0 = (blockIdx.x * 4 + wave) * 16;
  if (row0 >= M) return;
  int col0 = blockIdx.y * 64;
  int row = row0 + lm; if (row >= M) row = M - 1;
  const float* arow = A + (size_t)row * lda;
  v8f acc[4];
#pragma unroll
  for (int nt = 0; nt < 4; ++nt) {
    float bv = bias[col0 + nt*16 + lm];
#pragma unroll
    for (int r = 0; r < 8; ++r) acc[nt][r] = bv;
  }
#pragma unroll
  for (int kb = 0; kb < KP; kb += 32) {
    v16h a = load_frag_f<KA>(arow, kb, lh);
#pragma unroll
    for (int nt = 0; nt < 4; ++nt) {
      v16h b = load_frag_h(Wt + (size_t)(col0 + nt*16 + lm) * KP, kb, lh);
      acc[nt] = wmma32(a, b, acc[nt]);
    }
  }
#pragma unroll
  for (int nt = 0; nt < 4; ++nt)
#pragma unroll
    for (int r = 0; r < 8; ++r) {
      float v = acc[nt][r];
      if (act) v = leakyf(v);
      C[(size_t)(row0 + r + 8*lh) * ldc + col0 + nt*16 + lm] = (OutT)v;
    }
}

// --- fused per-edge MLP (phi): gather concat -> L1(256->64) -> leaky ->
//     L2(64->64); epilogue: f16 edge feats + f32 atomic segment-sum ---------
__global__ __launch_bounds__(128) void phi_fused(
    const half_t* __restrict__ xn, const half_t* __restrict__ ebuf,
    const half_t* __restrict__ gga, const int* __restrict__ eidx,
    const half_t* __restrict__ wt1, const float* __restrict__ b1,
    const half_t* __restrict__ wt2, const float* __restrict__ b2,
    half_t* __restrict__ e_out, float* __restrict__ agg)
{
  __shared__ half_t hid[4][16][72];   // stride 72 halves: 16B-aligned rows, conflict-free
  int lane = threadIdx.x & 31, wave = threadIdx.x >> 5;
  int lm = lane & 15, lh = lane >> 4;
  int j0 = (blockIdx.x * 4 + wave) * 16;
  int j  = j0 + lm;
  int g  = j / NEDGE;
  int le = j - g * NEDGE;
  int srcN = eidx[le]         + g * NNODE;
  int dstN = eidx[NEDGE + le] + g * NNODE;
  const half_t* srcs[4] = {
      xn   + (size_t)dstN * DD,               // x_i = x[dst]
      xn   + (size_t)srcN * DD,               // x_j = x[src]
      ebuf + (size_t)j * DD,                  // edge feats (flat-reshape quirk)
      gga  + (size_t)(j & (GNUM - 1)) * DD }; // gga[j % G] quirk

  v8f acc[4];
#pragma unroll
  for (int nt = 0; nt < 4; ++nt) {
    float bv = b1[nt*16 + lm];
#pragma unroll
    for (int r = 0; r < 8; ++r) acc[nt][r] = bv;
  }
#pragma unroll
  for (int c = 0; c < 8; ++c) {               // K = 256 = 8 chunks of 32
    v16h a = load_frag_h(srcs[c >> 1], (c & 1) * 32, lh);
#pragma unroll
    for (int nt = 0; nt < 4; ++nt) {
      v16h b = load_frag_h(wt1 + (size_t)(nt*16 + lm) * 256, c * 32, lh);
      acc[nt] = wmma32(a, b, acc[nt]);
    }
  }
#pragma unroll
  for (int nt = 0; nt < 4; ++nt)
#pragma unroll
    for (int r = 0; r < 8; ++r)
      hid[wave][r + 8*lh][nt*16 + lm] = (half_t)leakyf(acc[nt][r]);
  __syncthreads();

  v8f acc2[4];
#pragma unroll
  for (int nt = 0; nt < 4; ++nt) {
    float bv = b2[nt*16 + lm];
#pragma unroll
    for (int r = 0; r < 8; ++r) acc2[nt][r] = bv;
  }
  const half_t* hrow = &hid[wave][lm][0];
#pragma unroll
  for (int kc = 0; kc < 2; ++kc) {
    v16h a = load_frag_h(hrow, kc * 32, lh);
#pragma unroll
    for (int nt = 0; nt < 4; ++nt) {
      v16h b = load_frag_h(wt2 + (size_t)(nt*16 + lm) * 64, kc * 32, lh);
      acc2[nt] = wmma32(a, b, acc2[nt]);
    }
  }
#pragma unroll
  for (int r = 0; r < 8; ++r) {
    int jr = j0 + r + 8*lh;
    int gr = jr / NEDGE;
    int dr = eidx[NEDGE + (jr - gr * NEDGE)] + gr * NNODE;
#pragma unroll
    for (int nt = 0; nt < 4; ++nt) {
      float v = acc2[nt][r];
      e_out[(size_t)jr * DD + nt*16 + lm] = (half_t)v;
      atomicAdd(agg + (size_t)dr * DD + nt*16 + lm, v);   // segment_sum
    }
  }
}

// --- fused per-node MLP (gamma): [xn|agg|gga] -> L1(192->64) -> leaky ->
//     L2(64->64); epilogue: f16 node feats + per-graph mean pool ------------
__global__ __launch_bounds__(128) void gamma_fused(
    const half_t* __restrict__ xn, const float* __restrict__ agg,
    const half_t* __restrict__ gga,
    const half_t* __restrict__ wt1, const float* __restrict__ b1,
    const half_t* __restrict__ wt2, const float* __restrict__ b2,
    half_t* __restrict__ xn_out, float* __restrict__ pooled)
{
  __shared__ half_t hid[4][16][72];
  int lane = threadIdx.x & 31, wave = threadIdx.x >> 5;
  int lm = lane & 15, lh = lane >> 4;
  int tile = blockIdx.x * 4 + wave;           // tile == graph id (NN == 16)
  int n0 = tile * 16;
  int n  = n0 + lm;
  const half_t* xrow = xn  + (size_t)n * DD;
  const float*  arow = agg + (size_t)n * DD;
  const half_t* grow = gga + (size_t)(n & (GNUM - 1)) * DD;  // gga[n % G] quirk

  v8f acc[4];
#pragma unroll
  for (int nt = 0; nt < 4; ++nt) {
    float bv = b1[nt*16 + lm];
#pragma unroll
    for (int r = 0; r < 8; ++r) acc[nt][r] = bv;
  }
#pragma unroll
  for (int c = 0; c < 6; ++c) {               // K = 192 = 6 chunks
    v16h a;
    if (c < 2)      a = load_frag_h(xrow, (c & 1) * 32, lh);
    else if (c < 4) a = load_frag_f<64>(arow, (c & 1) * 32, lh);
    else            a = load_frag_h(grow, (c & 1) * 32, lh);
#pragma unroll
    for (int nt = 0; nt < 4; ++nt) {
      v16h b = load_frag_h(wt1 + (size_t)(nt*16 + lm) * 192, c * 32, lh);
      acc[nt] = wmma32(a, b, acc[nt]);
    }
  }
#pragma unroll
  for (int nt = 0; nt < 4; ++nt)
#pragma unroll
    for (int r = 0; r < 8; ++r)
      hid[wave][r + 8*lh][nt*16 + lm] = (half_t)leakyf(acc[nt][r]);
  __syncthreads();

  v8f acc2[4];
#pragma unroll
  for (int nt = 0; nt < 4; ++nt) {
    float bv = b2[nt*16 + lm];
#pragma unroll
    for (int r = 0; r < 8; ++r) acc2[nt][r] = bv;
  }
  const half_t* hrow = &hid[wave][lm][0];
#pragma unroll
  for (int kc = 0; kc < 2; ++kc) {
    v16h a = load_frag_h(hrow, kc * 32, lh);
#pragma unroll
    for (int nt = 0; nt < 4; ++nt) {
      v16h b = load_frag_h(wt2 + (size_t)(nt*16 + lm) * 64, kc * 32, lh);
      acc2[nt] = wmma32(a, b, acc2[nt]);
    }
  }
#pragma unroll
  for (int nt = 0; nt < 4; ++nt) {
    float psum = 0.f;
#pragma unroll
    for (int r = 0; r < 8; ++r) {
      float v = acc2[nt][r];
      xn_out[(size_t)(n0 + r + 8*lh) * DD + nt*16 + lm] = (half_t)v;
      psum += v;
    }
    float tot = psum + __shfl_xor(psum, 16, 32);   // combine M halves (wave32)
    if (lh == 0) pooled[(size_t)tile * DD + nt*16 + lm] = tot * (1.f / 16.f);
  }
}

// --- single-block persistent LSTM: 128 rows x 64 dims, 128 steps -----------
// 8 waves; each wave owns 16 rows: all 256 gate cols in 16 accumulators,
// cell state c lives in VGPRs, h round-trips through LDS (f32).
__global__ __launch_bounds__(256) void lstm_kernel(
    const half_t* __restrict__ seq, const half_t* __restrict__ wih,
    const half_t* __restrict__ whh, const float* __restrict__ bias,
    float* __restrict__ h_out)
{
  __shared__ float hs[128][68];
  int lane = threadIdx.x & 31, wave = threadIdx.x >> 5;
  int lm = lane & 15, lh = lane >> 4;
  int row0 = wave * 16;
  for (int i = threadIdx.x; i < 128 * 68; i += 256) (&hs[0][0])[i] = 0.f;
  v8f cst[4];
#pragma unroll
  for (int ct = 0; ct < 4; ++ct)
#pragma unroll
    for (int r = 0; r < 8; ++r) cst[ct][r] = 0.f;
  __syncthreads();

  for (int t = 0; t < TT; ++t) {
    v8f acc[16];
#pragma unroll
    for (int gt = 0; gt < 16; ++gt) {
      float bv = bias[gt*16 + lm];
#pragma unroll
      for (int r = 0; r < 8; ++r) acc[gt][r] = bv;
    }
    const half_t* xrow = seq + ((size_t)t * 128 + row0 + lm) * DD;
    const float*  hrow = &hs[row0 + lm][0];
#pragma unroll
    for (int kc = 0; kc < 2; ++kc) {
      v16h ax = load_frag_h(xrow, kc * 32, lh);
      v16h ah = load_frag_f<64>(hrow, kc * 32, lh);
#pragma unroll
      for (int gt = 0; gt < 16; ++gt) {
        v16h bx = load_frag_h(wih + (size_t)(gt*16 + lm) * DD, kc * 32, lh);
        acc[gt] = wmma32(ax, bx, acc[gt]);
        v16h bh = load_frag_h(whh + (size_t)(gt*16 + lm) * DD, kc * 32, lh);
        acc[gt] = wmma32(ah, bh, acc[gt]);
      }
    }
    __syncthreads();                       // all h reads done
#pragma unroll
    for (int ct = 0; ct < 4; ++ct)         // gate order i,f,g,o (torch)
#pragma unroll
      for (int r = 0; r < 8; ++r) {
        float iv = sigmf(acc[ct][r]);
        float fv = sigmf(acc[4 + ct][r]);
        float gv = tanhf(acc[8 + ct][r]);
        float ov = sigmf(acc[12 + ct][r]);
        float cn = fv * cst[ct][r] + iv * gv;
        cst[ct][r] = cn;
        hs[row0 + r + 8*lh][ct*16 + lm] = ov * tanhf(cn);
      }
    __syncthreads();                       // h visible to all waves
  }
#pragma unroll
  for (int ct = 0; ct < 4; ++ct)
#pragma unroll
    for (int r = 0; r < 8; ++r)
      h_out[(size_t)(row0 + r + 8*lh) * DD + ct*16 + lm] =
          hs[row0 + r + 8*lh][ct*16 + lm];
}

// --- regression head: gather follower rows, 64->64->32->5, 4*sigmoid+1 -----
__global__ __launch_bounds__(256) void head_kernel(
    const float* __restrict__ hfin, const int* __restrict__ fol,
    const float* __restrict__ w1, const float* __restrict__ b1,
    const float* __restrict__ w2, const float* __restrict__ b2,
    const float* __restrict__ w3, const float* __restrict__ b3,
    float* __restrict__ out)
{
  __shared__ float hs[8][64], h1[8][64], h2[8][32];
  int tid = threadIdx.x;
  for (int idx = tid; idx < 8 * 64; idx += 256) {
    int b = idx >> 6, d = idx & 63;
    hs[b][d] = hfin[(size_t)(b * NNODE + fol[b]) * DD + d];
  }
  __syncthreads();
  for (int idx = tid; idx < 8 * 64; idx += 256) {
    int b = idx >> 6, o = idx & 63;
    float s = b1[o];
    for (int k = 0; k < 64; ++k) s += hs[b][k] * w1[k * 64 + o];
    h1[b][o] = leakyf(s);
  }
  __syncthreads();
  for (int idx = tid; idx < 8 * 32; idx += 256) {
    int b = idx >> 5, o = idx & 31;
    float s = b2[o];
    for (int k = 0; k < 64; ++k) s += h1[b][k] * w2[k * 32 + o];
    h2[b][o] = leakyf(s);
  }
  __syncthreads();
  if (tid < 40) {
    int b = tid / 5, o = tid - b * 5;
    float s = b3[o];
    for (int k = 0; k < 32; ++k) s += h2[b][k] * w3[k * 5 + o];
    out[tid] = 4.f * sigmf(s) + 1.f;
  }
}

// ---------------------------------------------------------------------------
extern "C" void kernel_launch(void* const* d_in, const int* in_sizes, int n_in,
                              void* d_out, int out_size, void* d_ws, size_t ws_size,
                              hipStream_t stream)
{
  (void)in_sizes; (void)n_in; (void)out_size; (void)ws_size;
  // inputs in setup_inputs() dict order, params flattened in insertion order
  const float* x      = (const float*)d_in[0];
  const float* eattr  = (const float*)d_in[1];
  const float* gga_in = (const float*)d_in[2];
  const int*   eidx   = (const int*)d_in[3];
  const int*   fol    = (const int*)d_in[4];
  int pi = 5;
  const float* node_W = (const float*)d_in[pi++]; const float* node_b = (const float*)d_in[pi++];
  const float* edge_W = (const float*)d_in[pi++]; const float* edge_b = (const float*)d_in[pi++];
  const float* gga1_W = (const float*)d_in[pi++]; const float* gga1_b = (const float*)d_in[pi++];
  const float* gga2_W = (const float*)d_in[pi++]; const float* gga2_b = (const float*)d_in[pi++];
  struct MP { const float *p1W,*p1b,*p2W,*p2b,*g1W,*g1b,*g2W,*g2b,*q1W,*q1b,*q2W,*q2b; } mp[2];
  for (int l = 0; l < 2; ++l) {
    mp[l].p1W=(const float*)d_in[pi++]; mp[l].p1b=(const float*)d_in[pi++];
    mp[l].p2W=(const float*)d_in[pi++]; mp[l].p2b=(const float*)d_in[pi++];
    mp[l].g1W=(const float*)d_in[pi++]; mp[l].g1b=(const float*)d_in[pi++];
    mp[l].g2W=(const float*)d_in[pi++]; mp[l].g2b=(const float*)d_in[pi++];
    mp[l].q1W=(const float*)d_in[pi++]; mp[l].q1b=(const float*)d_in[pi++];
    mp[l].q2W=(const float*)d_in[pi++]; mp[l].q2b=(const float*)d_in[pi++];
  }
  const float* lstm_Wih = (const float*)d_in[pi++];
  const float* lstm_Whh = (const float*)d_in[pi++];
  const float* lstm_b   = (const float*)d_in[pi++];
  const float* r1W=(const float*)d_in[pi++]; const float* r1b=(const float*)d_in[pi++];
  const float* r2W=(const float*)d_in[pi++]; const float* r2b=(const float*)d_in[pi++];
  const float* r3W=(const float*)d_in[pi++]; const float* r3b=(const float*)d_in[pi++];

  // workspace carve-out
  char* ws = (char*)d_ws;
  size_t off = 0;
  auto alloc = [&](size_t bytes)->char* {
    char* p = ws + off; off += (bytes + 255) & ~(size_t)255; return p;
  };
  half_t* xn0   = (half_t*)alloc((size_t)NTOT*DD*2);
  half_t* xn1   = (half_t*)alloc((size_t)NTOT*DD*2);
  half_t* xn2   = (half_t*)alloc((size_t)NTOT*DD*2);
  half_t* e0    = (half_t*)alloc((size_t)ETOT*DD*2);
  half_t* e1    = (half_t*)alloc((size_t)ETOT*DD*2);
  float*  agg   = (float*) alloc((size_t)NTOT*DD*4);
  half_t* ggaA  = (half_t*)alloc((size_t)GNUM*DD*2);
  half_t* ggaB  = (half_t*)alloc((size_t)GNUM*DD*2);
  float*  pooled= (float*) alloc((size_t)GNUM*DD*4);
  float*  ggahid= (float*) alloc((size_t)GNUM*256*4);
  float*  pghid = (float*) alloc((size_t)GNUM*DD*4);
  float*  hout  = (float*) alloc((size_t)128*DD*4);
  half_t* node_wt = (half_t*)alloc(64*96*2);
  half_t* edge_wt = (half_t*)alloc(64*32*2);
  half_t* gga1_wt = (half_t*)alloc(256*32*2);
  half_t* gga2_wt = (half_t*)alloc(64*256*2);
  half_t *phi1_wt[2], *phi2_wt[2], *gam1_wt[2], *gam2_wt[2];
  for (int l = 0; l < 2; ++l) {
    phi1_wt[l] = (half_t*)alloc(64*256*2);
    phi2_wt[l] = (half_t*)alloc(64*64*2);
    gam1_wt[l] = (half_t*)alloc(64*192*2);
    gam2_wt[l] = (half_t*)alloc(64*64*2);
  }
  half_t* pg1_wt = (half_t*)alloc(64*64*2);
  half_t* pg2_wt = (half_t*)alloc(64*64*2);
  half_t* wih_wt = (half_t*)alloc(256*64*2);
  half_t* whh_wt = (half_t*)alloc(256*64*2);

  auto cvt = [&](const float* W, half_t* Wt, int K, int N, int Kp){
    int tot = N * Kp;
    wcvt_kernel<<<dim3((tot + 255) / 256), dim3(256), 0, stream>>>(W, Wt, K, N, Kp);
  };
  cvt(node_W, node_wt, 90, 64, 96);
  cvt(edge_W, edge_wt,  4, 64, 32);
  cvt(gga1_W, gga1_wt, 32, 256, 32);
  cvt(gga2_W, gga2_wt, 256, 64, 256);
  for (int l = 0; l < 2; ++l) {
    cvt(mp[l].p1W, phi1_wt[l], 256, 64, 256);
    cvt(mp[l].p2W, phi2_wt[l],  64, 64,  64);
    cvt(mp[l].g1W, gam1_wt[l], 192, 64, 192);
    cvt(mp[l].g2W, gam2_wt[l],  64, 64,  64);
  }
  cvt(mp[0].q1W, pg1_wt, 64, 64, 64);
  cvt(mp[0].q2W, pg2_wt, 64, 64, 64);
  cvt(lstm_Wih, wih_wt, 64, 256, 64);
  cvt(lstm_Whh, whh_wt, 64, 256, 64);

  // embeddings + gga MLP (templated shapes -> vectorized A loads)
  {
    dim3 blk(128);
    gemm_wmma<90,96,half_t><<<dim3(NTOT/64,1), blk, 0, stream>>>(x, 90, node_wt, node_b, xn0, 64, NTOT, 0);
    gemm_wmma<4,32,half_t><<<dim3(ETOT/64,1), blk, 0, stream>>>(eattr, 4, edge_wt, edge_b, e0, 64, ETOT, 0);
    gemm_wmma<32,32,float><<<dim3(GNUM/64,4), blk, 0, stream>>>(gga_in, 32, gga1_wt, gga1_b, ggahid, 256, GNUM, 1);
    gemm_wmma<256,256,half_t><<<dim3(GNUM/64,1), blk, 0, stream>>>(ggahid, 256, gga2_wt, gga2_b, ggaA, 64, GNUM, 0);
  }

  // MP layer 1
  hipMemsetAsync(agg, 0, (size_t)NTOT*DD*4, stream);
  phi_fused<<<dim3(ETOT/64), dim3(128), 0, stream>>>(
      xn0, e0, ggaA, eidx, phi1_wt[0], mp[0].p1b, phi2_wt[0], mp[0].p2b, e1, agg);
  gamma_fused<<<dim3(NTOT/64), dim3(128), 0, stream>>>(
      xn0, agg, ggaA, gam1_wt[0], mp[0].g1b, gam2_wt[0], mp[0].g2b, xn1, pooled);
  gemm_wmma<64,64,float><<<dim3(GNUM/64,1), dim3(128), 0, stream>>>(
      pooled, 64, pg1_wt, mp[0].q1b, pghid, 64, GNUM, 1);
  gemm_wmma<64,64,half_t><<<dim3(GNUM/64,1), dim3(128), 0, stream>>>(
      pghid, 64, pg2_wt, mp[0].q2b, ggaB, 64, GNUM, 0);

  // MP layer 2 (its phi_global output is never consumed -> skipped)
  hipMemsetAsync(agg, 0, (size_t)NTOT*DD*4, stream);
  phi_fused<<<dim3(ETOT/64), dim3(128), 0, stream>>>(
      xn1, e1, ggaB, eidx, phi1_wt[1], mp[1].p1b, phi2_wt[1], mp[1].p2b, e0, agg);
  gamma_fused<<<dim3(NTOT/64), dim3(128), 0, stream>>>(
      xn1, agg, ggaB, gam1_wt[1], mp[1].g1b, gam2_wt[1], mp[1].g2b, xn2, pooled);

  // LSTM over T=128 (seq[t] == rows t*128..t*128+127 of xn2, flat quirk)
  lstm_kernel<<<dim3(1), dim3(256), 0, stream>>>(xn2, wih_wt, whh_wt, lstm_b, hout);

  // head
  head_kernel<<<dim3(1), dim3(256), 0, stream>>>(
      hout, fol, r1W, r1b, r2W, r2b, r3W, r3b, (float*)d_out);
}
